// MixNetDecode_21199958573589
// MI455X (gfx1250) — matrix operand order repaired and checked by reading, not compiled, not run on hardware
//
#include <hip/hip_runtime.h>

// ---------------------------------------------------------------------------
// CDNA5 (gfx1250) WMMA helpers: bf16 16x16x32, wave32.
// ---------------------------------------------------------------------------
typedef __attribute__((ext_vector_type(16))) __bf16 v16bf;
typedef __attribute__((ext_vector_type(8)))  float  v8f;
typedef __attribute__((ext_vector_type(4)))  unsigned v4u;
typedef __attribute__((ext_vector_type(8)))  int      v8i;
typedef __attribute__((ext_vector_type(4)))  int      v4i;

__device__ __forceinline__ unsigned bfbits(float f) {
  union { float f; unsigned u; } v; v.f = f;
  return (v.u + 0x7FFFu + ((v.u >> 16) & 1u)) >> 16;
}

__device__ __forceinline__ unsigned pack_bf2(float a, float b) {
  return (bfbits(a) & 0xFFFFu) | (bfbits(b) << 16);
}

// ISA 10.2: flat pointers into the LDS aperture carry the LDS byte offset in
// addr[31:0] -> truncation recovers the TDM lds_addr.
__device__ __forceinline__ unsigned lds_off(const void* p) {
  return (unsigned)(unsigned long long)p;
}

#if __has_builtin(__builtin_amdgcn_tensor_load_to_lds)
// Issue a TDM 2D tile load: rows x cols elements of 4B, row stride in elems.
// 1D loads: rows=0 (tile_dim1 unused), cols = total elements.
__device__ __forceinline__ void tdm_load_f32(const float* gsrc, unsigned ldsDst,
                                             unsigned cols, unsigned rows,
                                             unsigned rowStride) {
  unsigned long long ga = (unsigned long long)gsrc;
  v4u g0;
  g0.x = 1u;                                                   // count=1, user mode
  g0.y = ldsDst;                                               // lds_addr
  g0.z = (unsigned)ga;                                         // global_addr[31:0]
  g0.w = (unsigned)((ga >> 32) & 0x01FFFFFFull) | (2u << 30);  // [56:32] | type=2
  unsigned td1 = rows ? rows : 1u;
  v8i g1;
  g1[0] = (int)(2u << 16);                         // data_size = 4B
  g1[1] = (int)((cols & 0xFFFFu) << 16);           // tensor_dim0 lo16
  g1[2] = (int)((cols >> 16) | ((td1 & 0xFFFFu) << 16));  // dim0 hi16 | dim1 lo16
  g1[3] = (int)((cols & 0xFFFFu) << 16);           // tile_dim0 (16-bit)
  g1[4] = (int)(rows & 0xFFFFu);                   // tile_dim1 (0 = 1D), tile_dim2=0
  g1[5] = (int)rowStride;                          // tensor_dim0_stride lo32
  g1[6] = 0;
  g1[7] = 0;
  v4i z4 = {};
#if defined(__clang_major__) && (__clang_major__ >= 23)
  v8i z8 = {};
  __builtin_amdgcn_tensor_load_to_lds(g0, g1, z4, z4, z8, 0);
#else
  __builtin_amdgcn_tensor_load_to_lds(g0, g1, z4, z4, 0);
#endif
}
#endif

// A fragment: 16(M) x 32(K) bf16, row-major source, leading dim lda (halves).
// Each lane = one contiguous 32B run (k = g*16 + e).
__device__ __forceinline__ v16bf frag_a(const __bf16* a, int lda, int lane) {
  int g = lane >> 4, m = lane & 15;
  v16bf f;
#pragma unroll
  for (int e = 0; e < 16; ++e) {
    int k = g * 16 + e;
    f[e] = a[m * lda + k];
  }
  return f;
}

// B fragment from K-major storage: col n at n*ldk, lane reads 16 contiguous
// halves at + g*16 -> 2x ds_load_b128.
__device__ __forceinline__ v16bf frag_b_k(const __bf16* colmaj, int ldk, int lane) {
  int g = lane >> 4, n = lane & 15;
  v16bf f;
  __builtin_memcpy(&f, colmaj + n * ldk + g * 16, sizeof(f));
  return f;
}

#define WMMA_BF16(A, B, C) \
  __builtin_amdgcn_wmma_f32_16x16x32_bf16(false, (A), false, (B), (short)0, (C), false, false)

// C/D layout (32-bit 16x16): vgpr r, lanes 0-15 -> M=r, N=lane;
// lanes 16-31 -> M=r+8, N=lane-16.

// ---------------------------------------------------------------------------
// Fused 3-layer 1x1-conv encoder. 4 waves, 16 spatial positions per block.
// Activation tile (C rows x 16 floats, row stride HW) streamed via TDM.
// ---------------------------------------------------------------------------
#define MAXC 256

__global__ void encode_kernel(const float* __restrict__ in,           // (B,C,HW)
                              const float* __restrict__ w1, const float* __restrict__ b1,  // (64,C)
                              const float* __restrict__ w2, const float* __restrict__ b2,  // (32,64)
                              const float* __restrict__ w3, const float* __restrict__ b3,  // (1,32)
                              float* __restrict__ feat,               // (B,HW)
                              int C, int HW) {
  __shared__ __align__(32) __bf16 sW1[64 * MAXC];
  __shared__ __align__(32) __bf16 sW2[32 * 64];
  __shared__ __align__(32) __bf16 sW3[16 * 32];
  __shared__ __align__(32) __bf16 sActT[16 * (MAXC + 8)];  // [n][k], ldk = C+8
  __shared__ __align__(32) __bf16 sH1T[16 * 72];           // [n][m], ldk = 72
  __shared__ __align__(32) __bf16 sH2T[16 * 40];           // [n][m], ldk = 40
#if __has_builtin(__builtin_amdgcn_tensor_load_to_lds)
  __shared__ __align__(16) float sAct32[MAXC * 16];        // TDM destination [c][n]
#endif

  int Cp = C + 8;
  int b = blockIdx.y;
  int n0 = blockIdx.x * 16;
  int tid = threadIdx.x;
  int lane = tid & 31, wave = tid >> 5;
  int g = lane >> 4, nl = lane & 15;

#if __has_builtin(__builtin_amdgcn_tensor_load_to_lds)
  // --- Tensor Data Mover: DMA the (C x 16f32) strided tile into LDS ---------
  if (wave == 0) {
    tdm_load_f32(in + (size_t)b * C * HW + n0, lds_off(sAct32),
                 16u, (unsigned)C, (unsigned)HW);
  }
#endif

  // Weights (packed 2 bf16 per b32 store) -- overlaps the TDM transfer
  for (int i = tid; i < (64 * C) >> 1; i += 128)
    *(unsigned*)&sW1[2 * i] = pack_bf2(w1[2 * i], w1[2 * i + 1]);
  for (int i = tid; i < (32 * 64) >> 1; i += 128)
    *(unsigned*)&sW2[2 * i] = pack_bf2(w2[2 * i], w2[2 * i + 1]);
  for (int i = tid; i < (16 * 32) >> 1; i += 128) {
    float a0 = (2 * i < 32) ? w3[2 * i] : 0.0f;
    float a1 = (2 * i + 1 < 32) ? w3[2 * i + 1] : 0.0f;
    *(unsigned*)&sW3[2 * i] = pack_bf2(a0, a1);
  }

#if __has_builtin(__builtin_amdgcn_tensor_load_to_lds)
#if __has_builtin(__builtin_amdgcn_s_wait_tensorcnt)
  if (wave == 0) __builtin_amdgcn_s_wait_tensorcnt(0);
#endif
  __syncthreads();  // publish TDM-written LDS to all waves
  // relu + fp32->bf16 transpose to K-major [n][c], LDS -> LDS
  for (int i = tid; i < (C >> 1) * 16; i += 128) {
    int kp = i >> 4, n = i & 15;
    int c0 = kp << 1;
    float v0 = fmaxf(sAct32[c0 * 16 + n], 0.0f);
    float v1 = fmaxf(sAct32[(c0 + 1) * 16 + n], 0.0f);
    *(unsigned*)&sActT[n * Cp + c0] = pack_bf2(v0, v1);
  }
#else
  // Fallback: direct strided global reads
  const float* src = in + (size_t)b * C * HW + n0;
  for (int i = tid; i < (C >> 1) * 16; i += 128) {
    int kp = i >> 4, n = i & 15;
    int c0 = kp << 1;
    float v0 = fmaxf(src[(size_t)c0 * HW + n], 0.0f);
    float v1 = fmaxf(src[(size_t)(c0 + 1) * HW + n], 0.0f);
    *(unsigned*)&sActT[n * Cp + c0] = pack_bf2(v0, v1);
  }
#endif
  __syncthreads();

  // Layer 1: [64 x C] * [C x 16]; wave w owns rows 16w..16w+15
  {
    v8f acc = {};
    for (int kt = 0; kt < C; kt += 32) {
      v16bf a = frag_a(sW1 + wave * 16 * C + kt, C, lane);
      v16bf bb = frag_b_k(sActT + kt, Cp, lane);
      acc = WMMA_BF16(a, bb, acc);
    }
#pragma unroll
    for (int r = 0; r < 8; r += 2) {
      int m = wave * 16 + g * 8 + r;  // even
      float u0 = fmaxf(acc[r] + b1[m], 0.0f);
      float u1 = fmaxf(acc[r + 1] + b1[m + 1], 0.0f);
      *(unsigned*)&sH1T[nl * 72 + m] = pack_bf2(u0, u1);
    }
  }
  __syncthreads();

  // Layer 2: [32 x 64] * [64 x 16]; waves 0,1
  if (wave < 2) {
    v8f acc = {};
    for (int kt = 0; kt < 64; kt += 32) {
      v16bf a = frag_a(sW2 + wave * 16 * 64 + kt, 64, lane);
      v16bf bb = frag_b_k(sH1T + kt, 72, lane);
      acc = WMMA_BF16(a, bb, acc);
    }
#pragma unroll
    for (int r = 0; r < 8; r += 2) {
      int m = wave * 16 + g * 8 + r;
      float u0 = fmaxf(acc[r] + b2[m], 0.0f);
      float u1 = fmaxf(acc[r + 1] + b2[m + 1], 0.0f);
      *(unsigned*)&sH2T[nl * 40 + m] = pack_bf2(u0, u1);
    }
  }
  __syncthreads();

  // Layer 3: [1(pad 16) x 32] * [32 x 16]; wave 0, row 0 valid
  if (wave == 0) {
    v8f acc = {};
    v16bf a = frag_a(sW3, 32, lane);
    v16bf bb = frag_b_k(sH2T, 40, lane);
    acc = WMMA_BF16(a, bb, acc);
    if (lane < 16) {  // M=0 lives in vgpr0 of lanes 0-15
      feat[(size_t)b * HW + n0 + lane] = fmaxf(acc[0] + b3[0], 0.0f);
    }
  }
}

// ---------------------------------------------------------------------------
// Cost volume + decoder matmul + softmax-expectation (both directions).
// Decoder weights (D*D contiguous fp32) streamed via a 1D TDM tile.
// ---------------------------------------------------------------------------
#define MAXD 96

__global__ void cost_dec_kernel(const float* __restrict__ fx, const float* __restrict__ fy,  // (B,HW)
                                const float* __restrict__ dw, const float* __restrict__ db,  // (D,D),(D)
                                float* __restrict__ dispx, float* __restrict__ dispy,        // (B,HW)
                                int D, int H, int W) {
  int Dp = (D + 31) & ~31;            // K padded to 32
  int Dpp = Dp + 8;                   // padded K-major pitch
  int mtiles = (D + 15) >> 4;
  int Mp = mtiles * 16;

  __shared__ __align__(32) __bf16 sDW[MAXD * MAXD];          // (Mp x Dp) row-major
  __shared__ __align__(32) __bf16 sCostT[16 * (MAXD + 8)];   // [n][d], ldk = Dpp
  __shared__ float sOut[MAXD * 16];
#if __has_builtin(__builtin_amdgcn_tensor_load_to_lds)
  __shared__ __align__(16) float sDW32[MAXD * MAXD];         // TDM destination
#endif

  int b = blockIdx.z, h = blockIdx.y, w0 = blockIdx.x * 16;
  int tid = threadIdx.x, lane = tid & 31, wave = tid >> 5;

#if __has_builtin(__builtin_amdgcn_tensor_load_to_lds)
  // --- TDM: DMA the contiguous (D*D f32) decoder matrix into LDS -----------
  if (wave == 0) {
    tdm_load_f32(dw, lds_off(sDW32), (unsigned)(D * D), 0u, (unsigned)(D * D));
#if __has_builtin(__builtin_amdgcn_s_wait_tensorcnt)
    __builtin_amdgcn_s_wait_tensorcnt(0);
#endif
  }
  __syncthreads();  // publish TDM-written LDS to all waves
  for (int i = tid; i < (Mp * Dp) >> 1; i += 128) {
    int j = 2 * i;
    int e = j / Dp, d = j % Dp;
    float a0 = (e < D && d < D) ? sDW32[e * D + d] : 0.0f;
    float a1 = (e < D && d + 1 < D) ? sDW32[e * D + d + 1] : 0.0f;
    *(unsigned*)&sDW[j] = pack_bf2(a0, a1);
  }
#else
  for (int i = tid; i < (Mp * Dp) >> 1; i += 128) {
    int j = 2 * i;
    int e = j / Dp, d = j % Dp;
    float a0 = (e < D && d < D) ? dw[e * D + d] : 0.0f;
    float a1 = (e < D && d + 1 < D) ? dw[e * D + d + 1] : 0.0f;
    *(unsigned*)&sDW[j] = pack_bf2(a0, a1);
  }
#endif
  const float* rowx = fx + ((size_t)b * H + h) * W;
  const float* rowy = fy + ((size_t)b * H + h) * W;
  __syncthreads();

  for (int dir = 0; dir < 2; ++dir) {
    for (int i = tid; i < (Dp >> 1) * 16; i += 128) {
      int kp = i >> 4, n = i & 15;
      int d0 = kp << 1;
      int w = w0 + n;
      float c0v = 0.0f, c1v = 0.0f;
      if (dir == 0) {
        if (d0 < D && w + d0 < W) c0v = rowx[w + d0] * rowy[w];
        if (d0 + 1 < D && w + d0 + 1 < W) c1v = rowx[w + d0 + 1] * rowy[w];
      } else {
        if (d0 < D && w - d0 >= 0) c0v = rowx[w] * rowy[w - d0];
        if (d0 + 1 < D && w - d0 - 1 >= 0) c1v = rowx[w] * rowy[w - d0 - 1];
      }
      *(unsigned*)&sCostT[n * Dpp + d0] = pack_bf2(c0v, c1v);
    }
    __syncthreads();

    for (int mt = wave; mt < mtiles; mt += 4) {
      v8f acc = {};
      for (int kt = 0; kt < Dp; kt += 32) {
        v16bf a = frag_a(sDW + mt * 16 * Dp + kt, Dp, lane);
        v16bf bb = frag_b_k(sCostT + kt, Dpp, lane);
        acc = WMMA_BF16(a, bb, acc);
      }
      int g = lane >> 4, n = lane & 15;
#pragma unroll
      for (int r = 0; r < 8; ++r) {
        int m = mt * 16 + r + g * 8;
        if (m < D) sOut[m * 16 + n] = acc[r] + db[m];
      }
    }
    __syncthreads();

    if (tid < 16) {
      float mx = -1e30f;
      for (int e = 0; e < D; ++e) mx = fmaxf(mx, sOut[e * 16 + tid]);
      float s = 0.0f, es = 0.0f;
      for (int e = 0; e < D; ++e) {
        float p = __expf(sOut[e * 16 + tid] - mx);
        s += p; es += p * (float)e;
      }
      float* dst = (dir == 0) ? dispx : dispy;
      dst[((size_t)b * H + h) * W + w0 + tid] = es / s;
    }
    __syncthreads();
  }
}

// ---------------------------------------------------------------------------
// Bilinear warp of img by disp along x; writes into a concat destination.
// ---------------------------------------------------------------------------
__global__ void warp_kernel(const float* __restrict__ img, const float* __restrict__ disp,
                            float* __restrict__ dst,
                            int C, int H, int W, float sign, int Ctot, int c0) {
  size_t idx = (size_t)blockIdx.x * blockDim.x + threadIdx.x;
  size_t total = (size_t)2 * C * H * W;
  if (idx >= total) return;
  int w = (int)(idx % W);
  int h = (int)((idx / W) % H);
  int c = (int)((idx / ((size_t)W * H)) % C);
  int b = (int)(idx / ((size_t)W * H * C));

  float d = disp[((size_t)b * H + h) * W + w];
  float xs = (float)w + sign * d;
  float xp = xs * (float)W / (float)(W - 1) - 0.5f;
  float yp = (float)h * (float)H / (float)(H - 1) - 0.5f;
  float x0 = floorf(xp), y0 = floorf(yp);
  int x0i = (int)x0, y0i = (int)y0;
  const float* ib = img + ((size_t)b * C + c) * H * W;
  float out = 0.0f;
#pragma unroll
  for (int dy = 0; dy < 2; ++dy) {
#pragma unroll
    for (int dx = 0; dx < 2; ++dx) {
      int xi = x0i + dx, yi = y0i + dy;
      bool valid = (xi >= 0) & (xi < W) & (yi >= 0) & (yi < H);
      int xc = xi < 0 ? 0 : (xi > W - 1 ? W - 1 : xi);
      int yc = yi < 0 ? 0 : (yi > H - 1 ? H - 1 : yi);
      float v = ib[(size_t)yc * W + xc];
      float wx = dx ? (xp - x0) : (x0 + 1.0f - xp);
      float wy = dy ? (yp - y0) : (y0 + 1.0f - yp);
      out += valid ? v * wx * wy : 0.0f;
    }
  }
  dst[((size_t)b * Ctot + c0 + c) * H * W + (size_t)h * W + w] = out;
}

// ---------------------------------------------------------------------------
// Channel-offset copy into a concat buffer (float4 / b128 path).
// ---------------------------------------------------------------------------
__global__ void copy_concat_kernel(const float4* __restrict__ src, float4* __restrict__ dst,
                                   int C, int Ctot, int c0, int HW4) {
  size_t idx = (size_t)blockIdx.x * blockDim.x + threadIdx.x;
  size_t total = (size_t)2 * C * HW4;
  if (idx >= total) return;
  int p = (int)(idx % HW4);
  int c = (int)((idx / HW4) % C);
  int b = (int)(idx / ((size_t)HW4 * C));
  dst[((size_t)b * Ctot + c0 + c) * HW4 + p] = src[idx];
}

// ---------------------------------------------------------------------------
// Transposed conv 4x4 stride 2 (lhs_dilation=2, pad 2), relu on input.
// Both x-phases computed per staged slab: 8 WMMA taps per 32-ci K-chunk,
// f32x2 coalesced stores, LDS-atomic stats reduction.
// ---------------------------------------------------------------------------
__global__ void upconv_kernel(const float* __restrict__ in,   // (B,Ci,H,W)
                              const float* __restrict__ wgt,  // (Ci,Co,4,4)
                              float* __restrict__ raw,        // (B,Co,2H,2W)
                              float* __restrict__ stats,      // (2*Co) sum, sumsq
                              int Ci, int Co, int H, int W) {
  __shared__ __align__(32) __bf16 sA[8][16 * 32];   // (px,j,i) taps, (co x ci) row-major
  __shared__ __align__(32) __bf16 sInT[136 * 40];   // [(jr*68+cc)][k], ldk = 40
  __shared__ float sSum[16], sSq[16];

  int b = blockIdx.z;
  int oy = blockIdx.y;
  int py = oy & 1, Y = oy >> 1;
  int X0b = blockIdx.x * 64;
  int tid = threadIdx.x, lane = tid & 31, wave = tid >> 5;
  int X0w = X0b + wave * 16;
  int Ho = 2 * H, Wo = 2 * W;
  int g = lane >> 4, nn = lane & 15;

  for (int cot = 0; cot < (Co >> 4); ++cot) {
    if (tid < 16) { sSum[tid] = 0.0f; sSq[tid] = 0.0f; }
    v8f acc0 = {}, acc1 = {};
    for (int kt = 0; kt < Ci; kt += 32) {
      // Stage weights: 8 taps (both px) x (16 co x 32 ci), packed pairs along ci
      for (int i = tid; i < 2048; i += 128) {
        int j2 = 2 * i;
        int tap = j2 >> 9, rem = j2 & 511;
        int m = rem >> 5, k0 = rem & 31;
        int pxt = tap >> 2, j = (tap >> 1) & 1, ii = tap & 1;
        int ky = py + 2 * j, kx = pxt + 2 * ii;
        size_t w0i = (((size_t)(kt + k0) * Co + (cot * 16 + m)) * 4 + (3 - ky)) * 4 + (3 - kx);
        size_t w1i = w0i + (size_t)Co * 16;  // next ci
        *(unsigned*)&sA[tap][m * 32 + k0] = pack_bf2(wgt[w0i], wgt[w1i]);
      }
      // Stage input slab K-major: 2 rows x 68 cols x 32 ci, relu + zero borders
      for (int i = tid; i < 2 * 68 * 16; i += 128) {
        int cc = i % 68;
        int t = i / 68;            // 0..31
        int kp = t & 15, jr = t >> 4;
        int k0 = kp << 1;
        int iy = Y + py + jr - 1;
        int ix = X0b + cc - 1;
        float v0 = 0.0f, v1 = 0.0f;
        if (iy >= 0 && iy < H && ix >= 0 && ix < W) {
          const float* p = &in[(((size_t)b * Ci + kt + k0) * H + iy) * W + ix];
          __builtin_prefetch(p + W, 0, 1);  // global_prefetch_b8
          v0 = fmaxf(p[0], 0.0f);
          v1 = fmaxf(p[(size_t)H * W], 0.0f);
        }
        *(unsigned*)&sInT[(jr * 68 + cc) * 40 + k0] = pack_bf2(v0, v1);
      }
      __syncthreads();

      if (X0w < W) {  // wave-uniform guard: EXEC all-ones inside
#pragma unroll
        for (int j = 0; j < 2; ++j) {
#pragma unroll
          for (int ii = 0; ii < 2; ++ii) {
            v16bf a0 = frag_a(sA[0 * 4 + j * 2 + ii], 32, lane);
            v16bf b0 = frag_b_k(sInT + (j * 68 + wave * 16 + 0 + ii) * 40, 40, lane);
            acc0 = WMMA_BF16(a0, b0, acc0);
            v16bf a1 = frag_a(sA[1 * 4 + j * 2 + ii], 32, lane);
            v16bf b1 = frag_b_k(sInT + (j * 68 + wave * 16 + 1 + ii) * 40, 40, lane);
            acc1 = WMMA_BF16(a1, b1, acc1);
          }
        }
      }
      __syncthreads();
    }
    if (X0w < W) {
      int xo2 = 2 * (X0w + nn);     // even: px=0 and px=1 are adjacent
#pragma unroll
      for (int r = 0; r < 8; ++r) {
        int cr = r + g * 8;         // channel within 16-wide tile
        int co = cot * 16 + cr;
        float v0 = acc0[r], v1 = acc1[r];
        float2 p2; p2.x = v0; p2.y = v1;
        *(float2*)&raw[(((size_t)b * Co + co) * Ho + oy) * Wo + xo2] = p2;
        atomicAdd(&sSum[cr], v0 + v1);      // ds_add_f32
        atomicAdd(&sSq[cr], v0 * v0 + v1 * v1);
      }
    }
    __syncthreads();
    if (tid < 16) {
      atomicAdd(&stats[cot * 16 + tid], sSum[tid]);
      atomicAdd(&stats[Co + cot * 16 + tid], sSq[tid]);
    }
    __syncthreads();
  }
}

__global__ void norm_kernel(const float4* __restrict__ raw, const float* __restrict__ stats,
                            const float* __restrict__ gam, const float* __restrict__ bet,
                            float4* __restrict__ dst, int Co, int HWo4) {
  size_t idx = (size_t)blockIdx.x * blockDim.x + threadIdx.x;
  size_t total = (size_t)2 * Co * HWo4;
  if (idx >= total) return;
  int co = (int)((idx / HWo4) % Co);
  float cnt = 2.0f * 4.0f * (float)HWo4;
  float m = stats[co] / cnt;
  float var = stats[Co + co] / cnt - m * m;
  float sc = gam[co] * rsqrtf(var + 1e-5f);
  float sh = bet[co] - m * sc;
  float4 v = raw[idx];
  float4 o;
  o.x = v.x * sc + sh; o.y = v.y * sc + sh;
  o.z = v.z * sc + sh; o.w = v.w * sc + sh;
  dst[idx] = o;
}

// ---------------------------------------------------------------------------
// Host orchestration.
// Input order: 0..9 images, 10..17 pa8, 18..25 pa4, 26..33 pa2, 34..41 pa1,
// 42..59 up (x4:w,g,b  y4  x2  y2  x1  y1).
// Output: xm1 | ym1 | xm2 | ym2 (fp32, flat).
// ---------------------------------------------------------------------------
extern "C" void kernel_launch(void* const* d_in, const int* in_sizes, int n_in,
                              void* d_out, int out_size, void* d_ws, size_t ws_size,
                              hipStream_t stream) {
  (void)in_sizes; (void)n_in; (void)out_size; (void)ws_size;
  auto F = [&](int i) { return (const float*)d_in[i]; };

  float* wsf = (float*)d_ws;
  size_t off = 0;
  auto alloc = [&](size_t n) { float* p = wsf + off; off += n; return p; };

  float* featx = alloc(131072);
  float* featy = alloc(131072);
  float* dx    = alloc(131072);
  float* dy    = alloc(131072);
  float* xm8   = alloc((size_t)2 * 768 * 1024);
  float* ym8   = alloc((size_t)2 * 768 * 1024);
  float* xd4   = alloc((size_t)2 * 128 * 4096);
  float* yd4   = alloc((size_t)2 * 128 * 4096);
  float* xm4   = alloc((size_t)2 * 384 * 4096);
  float* ym4   = alloc((size_t)2 * 384 * 4096);
  float* xd2   = alloc((size_t)2 * 64 * 16384);
  float* yd2   = alloc((size_t)2 * 64 * 16384);
  float* xd1   = alloc((size_t)2 * 32 * 65536);
  float* yd1   = alloc((size_t)2 * 32 * 65536);
  float* raw   = alloc((size_t)2 * 32 * 65536);   // largest raw up output
  float* stats = alloc(256);

  float* outp = (float*)d_out;
  float* xm1 = outp;
  float* ym1 = outp + (size_t)2 * 96 * 65536;
  float* xm2 = ym1 + (size_t)2 * 96 * 65536;
  float* ym2 = xm2 + (size_t)2 * 192 * 16384;

  auto run_scale = [&](const float* x, const float* y, const float* xd, const float* yd,
                       int C, int H, int W, int D, int pa,
                       float* xm, float* ym) {
    int HW = H * W;
    encode_kernel<<<dim3(HW / 16, 2), 128, 0, stream>>>(
        xd, F(pa), F(pa + 1), F(pa + 2), F(pa + 3), F(pa + 4), F(pa + 5), featx, C, HW);
    encode_kernel<<<dim3(HW / 16, 2), 128, 0, stream>>>(
        yd, F(pa), F(pa + 1), F(pa + 2), F(pa + 3), F(pa + 4), F(pa + 5), featy, C, HW);
    cost_dec_kernel<<<dim3(W / 16, H, 2), 128, 0, stream>>>(
        featx, featy, F(pa + 6), F(pa + 7), dx, dy, D, H, W);
    size_t nw = (size_t)2 * C * HW;
    warp_kernel<<<(int)((nw + 255) / 256), 256, 0, stream>>>(yd, dx, xm, C, H, W, -1.0f, 3 * C, 2 * C);
    warp_kernel<<<(int)((nw + 255) / 256), 256, 0, stream>>>(xd, dy, ym, C, H, W, +1.0f, 3 * C, C);
    int HW4 = HW / 4;
    size_t nc = (size_t)2 * C * HW4;
    copy_concat_kernel<<<(int)((nc + 255) / 256), 256, 0, stream>>>(
        (const float4*)x,  (float4*)xm, C, 3 * C, 0,     HW4);
    copy_concat_kernel<<<(int)((nc + 255) / 256), 256, 0, stream>>>(
        (const float4*)xd, (float4*)xm, C, 3 * C, C,     HW4);
    copy_concat_kernel<<<(int)((nc + 255) / 256), 256, 0, stream>>>(
        (const float4*)y,  (float4*)ym, C, 3 * C, 0,     HW4);
    copy_concat_kernel<<<(int)((nc + 255) / 256), 256, 0, stream>>>(
        (const float4*)yd, (float4*)ym, C, 3 * C, 2 * C, HW4);
  };

  auto run_up = [&](const float* src, int upIdx, int Ci, int Co, int H, int W, float* dst) {
    hipMemsetAsync(stats, 0, 2 * Co * sizeof(float), stream);
    upconv_kernel<<<dim3((W + 63) / 64, 2 * H, 2), 128, 0, stream>>>(
        src, F(upIdx), raw, stats, Ci, Co, H, W);
    int HWo4 = H * W;  // (4*H*W)/4
    size_t total = (size_t)2 * Co * HWo4;
    norm_kernel<<<(int)((total + 255) / 256), 256, 0, stream>>>(
        (const float4*)raw, stats, F(upIdx + 1), F(upIdx + 2), (float4*)dst, Co, HWo4);
  };

  // Scale 1/8: C=256, 32x32, D=12
  run_scale(F(0), F(1), F(2), F(3), 256, 32, 32, 12, 10, xm8, ym8);
  run_up(xm8, 42, 768, 128, 32, 32, xd4);   // up.x4
  run_up(ym8, 45, 768, 128, 32, 32, yd4);   // up.y4

  // Scale 1/4: C=128, 64x64, D=24
  run_scale(F(4), F(5), xd4, yd4, 128, 64, 64, 24, 18, xm4, ym4);
  run_up(xm4, 48, 384, 64, 64, 64, xd2);    // up.x2
  run_up(ym4, 51, 384, 64, 64, 64, yd2);    // up.y2

  // Scale 1/2: C=64, 128x128, D=48 -> xm2/ym2 are outputs
  run_scale(F(6), F(7), xd2, yd2, 64, 128, 128, 48, 26, xm2, ym2);
  run_up(xm2, 54, 192, 32, 128, 128, xd1);  // up.x1
  run_up(ym2, 57, 192, 32, 128, 128, yd1);  // up.y1

  // Scale 1/1: C=32, 256x256, D=96 -> xm1/ym1 are outputs
  run_scale(F(8), F(9), xd1, yd1, 32, 256, 256, 96, 34, xm1, ym1);
}
